// VSSM_1889785610536
// MI455X (gfx1250) — matrix operand order repaired and checked by reading, not compiled, not run on hardware
//
#include <hip/hip_runtime.h>
#include <cmath>
#include <cstdint>

typedef __attribute__((ext_vector_type(16))) _Float16 v16h;
typedef __attribute__((ext_vector_type(8)))  _Float16 v8h;
typedef __attribute__((ext_vector_type(8)))  float    v8f;
typedef int v4i __attribute__((vector_size(16)));    // matches builtin param type

#define B__  2
#define DM   96
#define HH   64
#define WW   64
#define LL   4096
#define KD   4      // scan directions
#define NS   8      // d_state
#define RR   6      // dt_rank
#define CC   22     // R + 2N
#define CHUNK 128

#if defined(__gfx1250__) && \
    __has_builtin(__builtin_amdgcn_global_load_async_to_lds_b128) && \
    __has_builtin(__builtin_amdgcn_s_wait_asynccnt)
#define USE_ASYNC_LDS 1
#else
#define USE_ASYNC_LDS 0
#endif

// ---------------- kernel 0: zero the output (harness poisons it) ----------------
__global__ void vssm_zero(float* __restrict__ out, int n) {
  int i = blockIdx.x * blockDim.x + threadIdx.x;
  if (i < n) out[i] = 0.f;
}

// ---------------- kernel 1: x_dbl[b,k] = W_k (22x96) @ xs[b,k] (96xL) via WMMA ---
// grid: B*K*(L/128) blocks of 32 threads (one wave32). Cross-scan applied by
// index mapping while staging xs tiles into LDS (no materialized xs).
__global__ void vssm_proj(const float* __restrict__ x,
                          const float* __restrict__ xpw,
                          float* __restrict__ xdbl) {
  __shared__ __align__(32) _Float16 wlds[32 * 96];  // [c_padded][d], rows 22..31 = 0
  __shared__ __align__(32) _Float16 xlds[16 * 96];  // [j][d] xs tile (transposed!)

  const int nblk_per_bk = LL / 128;                 // 32
  const int bk     = blockIdx.x / nblk_per_bk;
  const int b      = bk / KD, k = bk % KD;
  const int lbase0 = (blockIdx.x % nblk_per_bk) * 128;

  const int lane = threadIdx.x;      // 0..31
  const int half = lane >> 4;        // 0: lanes 0-15, 1: lanes 16-31
  const int mrow = lane & 15;
  const int ncol = lane & 15;

  // stage direction-k projection weights (22x96), padded to 32 rows
  const float* Wk = xpw + (size_t)k * CC * DM;
  for (int i = lane; i < 32 * 96; i += 32) {
    int c = i / 96, d = i - c * 96;
    wlds[i] = (c < CC) ? (_Float16)Wk[c * DM + d] : (_Float16)0.f;
  }
  __syncthreads();

  const float* xb = x + (size_t)b * DM * LL;

  for (int t = 0; t < 8; ++t) {                     // 8 l-tiles of 16 -> 128 cols
    const int lbase = lbase0 + t * 16;
    __syncthreads();                                // protect xlds WAR across tiles
    for (int i = lane; i < 16 * 96; i += 32) {
      int j = i / 96, d = i - j * 96;
      int l = lbase + j;
      int pos;
      if      (k == 0) pos = l;
      else if (k == 1) pos = ((l & 63) << 6) + (l >> 6);
      else if (k == 2) pos = LL - 1 - l;
      else { int lr = LL - 1 - l; pos = ((lr & 63) << 6) + (lr >> 6); }
      xlds[i] = (_Float16)xb[(size_t)d * LL + pos];
    }
    __syncthreads();

    float* outbk = xdbl + (size_t)(b * KD + k) * CC * LL;
    for (int mt = 0; mt < 2; ++mt) {
      v8f acc = {};
#pragma unroll
      for (int kc = 0; kc < 3; ++kc) {              // contract D=96 in 3 x K=32
        const int kbase = kc * 32;
        // A (16x32 f16): per-lane K spans {kbase+half*8 .. +7} and {+16 more}
        const int arow = (mt * 16 + mrow) * 96 + kbase + half * 8;
        v8h alo = *(const v8h*)&wlds[arow];
        v8h ahi = *(const v8h*)&wlds[arow + 16];
        v16h a  = __builtin_shufflevector(alo, ahi, 0, 1, 2, 3, 4, 5, 6, 7,
                                          8, 9, 10, 11, 12, 13, 14, 15);
        // B (32x16 f16): per-lane K spans {kbase+half*8 .. +7} and {+16 more} at [ncol]
        v8h blo = *(const v8h*)&xlds[ncol * 96 + kbase + half * 8];
        v8h bhi = *(const v8h*)&xlds[ncol * 96 + kbase + half * 8 + 16];
        v16h bm = __builtin_shufflevector(blo, bhi, 0, 1, 2, 3, 4, 5, 6, 7,
                                          8, 9, 10, 11, 12, 13, 14, 15);
        acc = __builtin_amdgcn_wmma_f32_16x16x32_f16(
            /*neg_a=*/false, a, /*neg_b=*/false, bm,
            /*c_mod=*/(short)0, acc, /*reuse_a=*/false, /*reuse_b=*/false);
      }
      // C/D layout: VGPR v = row M=v (lanes 0-15) or M=v+8 (lanes 16-31)
#pragma unroll
      for (int v = 0; v < 8; ++v) {
        int c = mt * 16 + v + (half ? 8 : 0);
        if (c < CC) outbk[(size_t)c * LL + lbase + ncol] = acc[v];
      }
    }
  }
}

// ---------------- kernel 2: fused dt-proj + softplus + selective scan + merge ---
// grid: B*K blocks of 96 threads (3 waves); thread = channel d. The 22 shared
// x_dbl rows are staged per 128-column chunk through LDS (async-to-LDS path on
// gfx1250 when available). Scan state (8 floats) lives in registers.
// Cross-merge position == cross-scan gather position -> atomicAdd into out.
__global__ void vssm_scan(const float* __restrict__ x,
                          const float* __restrict__ xdbl,
                          const float* __restrict__ dtw,
                          const float* __restrict__ dtb,
                          const float* __restrict__ alog,
                          const float* __restrict__ dsv,
                          float* __restrict__ out) {
  __shared__ __align__(16) float cb[CC * CHUNK];    // 22 x 128 floats

  const int bk = blockIdx.x;
  const int b  = bk / KD, k = bk % KD;
  const int d  = threadIdx.x;                       // 0..95
  const int kd = k * DM + d;

  float A[NS], h[NS];
#pragma unroll
  for (int n = 0; n < NS; ++n) {
    A[n] = -expf(alog[(size_t)kd * NS + n]);
    h[n] = 0.f;
  }
  float w[RR];
#pragma unroll
  for (int r = 0; r < RR; ++r) w[r] = dtw[((size_t)k * DM + d) * RR + r];
  const float bias = dtb[kd];
  const float Dd   = dsv[kd];

  const float* xdb = xdbl + (size_t)bk * CC * LL;
  const float* xr  = x   + ((size_t)b * DM + d) * LL;
  float*       ob  = out + ((size_t)b * DM + d) * LL;

  for (int cs = 0; cs < LL; cs += CHUNK) {
    __syncthreads();
#if USE_ASYNC_LDS
    for (int i = threadIdx.x; i < (CC * CHUNK) / 4; i += DM) {
      int c = i / (CHUNK / 4), q = i - c * (CHUNK / 4);
      const float* g = xdb + (size_t)c * LL + cs + q * 4;
      float*       p = &cb[c * CHUNK + q * 4];
      __builtin_amdgcn_global_load_async_to_lds_b128((v4i*)g, (v4i*)p, 0, 0);
    }
    __builtin_amdgcn_s_wait_asynccnt(0);
#else
    for (int i = threadIdx.x; i < CC * CHUNK; i += DM) {
      int c = i / CHUNK, j = i - c * CHUNK;
      cb[i] = xdb[(size_t)c * LL + cs + j];
    }
    if (cs + CHUNK < LL)                            // hint next chunk into cache
      __builtin_prefetch(&xdb[cs + CHUNK], 0, 1);
#endif
    __syncthreads();

    for (int j = 0; j < CHUNK; ++j) {
      const int l = cs + j;
      int pos;                                      // gather pos == merge pos
      if      (k == 0) pos = l;
      else if (k == 1) pos = ((l & 63) << 6) + (l >> 6);
      else if (k == 2) pos = LL - 1 - l;
      else { int lr = LL - 1 - l; pos = ((lr & 63) << 6) + (lr >> 6); }

      const float u = xr[pos];

      float dtv = bias;
#pragma unroll
      for (int r = 0; r < RR; ++r) dtv += w[r] * cb[r * CHUNK + j];
      const float delta = (dtv > 20.f) ? dtv : log1pf(expf(dtv));

      float y = Dd * u;
#pragma unroll
      for (int n = 0; n < NS; ++n) {
        const float da = expf(delta * A[n]);
        h[n] = da * h[n] + (delta * cb[(RR + n) * CHUNK + j]) * u;
        y += h[n] * cb[(RR + NS + n) * CHUNK + j];
      }
      atomicAdd(&ob[pos], y);
    }
  }
}

extern "C" void kernel_launch(void* const* d_in, const int* in_sizes, int n_in,
                              void* d_out, int out_size, void* d_ws, size_t ws_size,
                              hipStream_t stream) {
  const float* x    = (const float*)d_in[0];   // (B, D, H, W)
  const float* xpw  = (const float*)d_in[1];   // (K, 22, D)
  const float* dtw  = (const float*)d_in[2];   // (K, D, R)
  const float* dtb  = (const float*)d_in[3];   // (K, D)
  const float* alog = (const float*)d_in[4];   // (K*D, N)
  const float* dsv  = (const float*)d_in[5];   // (K*D,)
  float* out  = (float*)d_out;                 // (B, D, H, W) fp32
  float* xdbl = (float*)d_ws;                  // B*K*22*L floats (~2.75 MB)

  const int total_out = B__ * DM * LL;
  vssm_zero<<<(total_out + 255) / 256, 256, 0, stream>>>(out, total_out);
  vssm_proj<<<B__ * KD * (LL / 128), 32, 0, stream>>>(x, xpw, xdbl);
  vssm_scan<<<B__ * KD, DM, 0, stream>>>(x, xdbl, dtw, dtb, alog, dsv, out);
}
